// CausalSelfAttention_70574902608356
// MI455X (gfx1250) — matrix operand order repaired and checked by reading, not compiled, not run on hardware
//
#include <hip/hip_runtime.h>
#include <hip/hip_bf16.h>

// CDNA5 / gfx1250 causal self-attention forward.
//  (0) f32 -> f16 conversion pass for x, w_qkv, w_proj
//  (1) QKV GEMM: 4-wave WG, 64x64 tile; weight tile async-staged to LDS
//      (double buffered, global_load_async_to_lds_b128 + s_wait_asynccnt),
//      f16 WMMA / f32 acc; scatter Q[bh][t][d], K[bh][t][d], V^T[bh][d][t].
//  (2) Flash attention: 4-wave WG, 64 queries; K/V 32-key tiles async-staged
//      to LDS, WMMA QK^T and PV, f32 online softmax, per-wave LDS P relayout.
//  (3) Projection GEMM, same structure as (1), f32 output + bias.

typedef __attribute__((ext_vector_type(16))) _Float16 v16h;
typedef __attribute__((ext_vector_type(8)))  _Float16 h8;
typedef __attribute__((ext_vector_type(8)))  float    v8f;

#define BB   4
#define TT   2048
#define CC   1024
#define HH   16
#define HD   64
#define NROW (BB*TT)   /* 8192 */

#define KPAD 80   /* halves per K-tile LDS row (64 + 16 pad) */
#define VPAD 40   /* halves per V-tile LDS row (32 + 8 pad)  */
#define BPAD 40   /* halves per B-tile LDS row (32 + 8 pad)  */

static __device__ __forceinline__ v16h cat16(h8 lo, h8 hi) {
  return __builtin_shufflevector(lo, hi, 0,1,2,3,4,5,6,7,8,9,10,11,12,13,14,15);
}
static __device__ __forceinline__ h8 ld8(const _Float16* p) { return *(const h8*)p; }

static __device__ __forceinline__ v8f wmma(v16h a, v16h b, v8f c) {
  return __builtin_amdgcn_wmma_f32_16x16x32_f16(false, a, false, b, (short)0, c, false, false);
}

// Async global -> LDS copy of 16 bytes (gfx1250, tracked by ASYNCcnt).
// lds_off is the wave-relative LDS byte address (low 32 bits of a generic
// pointer to a __shared__ object).
static __device__ __forceinline__ void async_b128(unsigned lds_off, const void* g) {
  asm volatile("global_load_async_to_lds_b128 %0, %1, off"
               :: "v"(lds_off), "v"(g) : "memory");
}
static __device__ __forceinline__ void wait_async0() {
  asm volatile("s_wait_asynccnt 0x0" ::: "memory");
}
static __device__ __forceinline__ unsigned lds_addr(const void* p) {
  return (unsigned)(size_t)p;   // low 32 bits of generic LDS address = offset
}

// ---------------------------------------------------------------------------
// Kernel 0: f32 -> f16, 8 elements/thread
// ---------------------------------------------------------------------------
__global__ __launch_bounds__(256)
void cvt_kernel(const float* __restrict__ in, _Float16* __restrict__ out, int n)
{
  const int i = (blockIdx.x * 256 + threadIdx.x) * 8;
  if (i + 8 <= n) {
    v8f f = *(const v8f*)(in + i);
    *(h8*)(out + i) = __builtin_convertvector(f, h8);
  }
}

// ---------------------------------------------------------------------------
// Kernel 1: qkv = x @ w_qkv^T + b_qkv. WG = 128 threads (4 waves).
// Tile: M=64 (16 per wave) x N=64 shared. Weight tile (64x32 f16) staged in
// LDS via async loads, double buffered.
// ---------------------------------------------------------------------------
__global__ __launch_bounds__(128)
void qkv_kernel(const _Float16* __restrict__ xh, const _Float16* __restrict__ wh,
                const float* __restrict__ bias,
                _Float16* __restrict__ qws, _Float16* __restrict__ kws,
                _Float16* __restrict__ vwsT)
{
  __shared__ _Float16 bt[2][64 * BPAD];

  const int tid  = threadIdx.x;
  const int wv   = tid >> 5;
  const int lane = tid & 31;
  const int hl   = lane >> 4;
  const int l15  = lane & 15;
  const int n0   = blockIdx.x * 64;
  const int m0   = blockIdx.y * 64 + wv * 16;

  auto stage = [&](int k0, int buf) {
    const unsigned base = lds_addr(&bt[buf][0]);
#pragma unroll
    for (int i = 0; i < 2; ++i) {
      const int c  = tid + 128 * i;       // 256 chunks = 64 rows x 4 x 16B
      const int r  = c >> 2, ck = c & 3;
      async_b128(base + (unsigned)(r * BPAD + ck * 8) * 2,
                 wh + (size_t)(n0 + r) * CC + k0 + ck * 8);
    }
  };

  v8f acc[4] = {};
  const _Float16* arow = xh + (size_t)(m0 + l15) * CC;

  stage(0, 0);
  wait_async0();
  __syncthreads();

  for (int k0 = 0; k0 < CC; k0 += 32) {
    const int buf = (k0 >> 5) & 1;
    if (k0 + 32 < CC) stage(k0 + 32, buf ^ 1);

    const int cb = k0 + (hl ? 8 : 0);
    v16h a = cat16(ld8(arow + cb), ld8(arow + cb + 16));
#pragma unroll
    for (int s = 0; s < 4; ++s) {
      const _Float16* br = &bt[buf][(s * 16 + l15) * BPAD + (hl ? 8 : 0)];
      v16h b = cat16(ld8(br), ld8(br + 16));
      acc[s] = wmma(a, b, acc[s]);
    }
    wait_async0();
    __syncthreads();
  }

#pragma unroll
  for (int s = 0; s < 4; ++s) {
    const int n = n0 + s * 16 + l15;
    const float bv = bias[n];
    int c, kind;
    if (n < CC)        { c = n;        kind = 0; }
    else if (n < 2*CC) { c = n - CC;   kind = 1; }
    else               { c = n - 2*CC; kind = 2; }
    const int h = c >> 6, d = c & 63;
#pragma unroll
    for (int g = 0; g < 8; ++g) {
      const int m  = m0 + g + (hl ? 8 : 0);
      const int b_ = m >> 11, t = m & (TT - 1);
      const int bh = b_ * HH + h;
      const _Float16 val = (_Float16)(acc[s][g] + bv);
      if (kind == 0)      qws [((size_t)(bh*TT + t))*HD + d] = val;
      else if (kind == 1) kws [((size_t)(bh*TT + t))*HD + d] = val;
      else                vwsT[((size_t)(bh*HD + d))*TT + t] = val;
    }
  }
}

// ---------------------------------------------------------------------------
// Kernel 2: flash attention. WG = 128 threads; 64 queries (16/wave) share
// async-staged 32-key K/V tiles (double buffered).
// ---------------------------------------------------------------------------
__global__ __launch_bounds__(128)
void attn_kernel(const _Float16* __restrict__ qws, const _Float16* __restrict__ kws,
                 const _Float16* __restrict__ vwsT, _Float16* __restrict__ yws)
{
  __shared__ _Float16 kt[2][32 * KPAD];   // 32 kv rows x 64 d
  __shared__ _Float16 vt[2][64 * VPAD];   // 64 d rows x 32 kv
  __shared__ _Float16 pt[4][16 * 32];     // per-wave P tile

  const int tid  = threadIdx.x;
  const int wv   = tid >> 5;
  const int lane = tid & 31;
  const int hl   = lane >> 4;
  const int l15  = lane & 15;
  const int bh   = blockIdx.y;
  const int bq   = bh >> 4;
  const int hh   = bh & 15;
  const int qb   = blockIdx.x * 64;
  const int q0   = qb + wv * 16;
  const int jend = qb + 64;

  auto stagekv = [&](int j, int buf) {
    const unsigned kb = lds_addr(&kt[buf][0]);
    const unsigned vb = lds_addr(&vt[buf][0]);
#pragma unroll
    for (int i = 0; i < 2; ++i) {
      const int c = tid + 128 * i;        // 256 chunks each
      { const int r = c >> 3, ck = c & 7; // K: 32 rows x 8 x 16B
        async_b128(kb + (unsigned)(r * KPAD + ck * 8) * 2,
                   kws + (size_t)(bh*TT + j + r) * HD + ck * 8); }
      { const int r = c >> 2, ck = c & 3; // V^T: 64 rows x 4 x 16B
        async_b128(vb + (unsigned)(r * VPAD + ck * 8) * 2,
                   vwsT + (size_t)(bh*HD + r) * TT + j + ck * 8); }
    }
  };

  // Q as two A-operands (HD=64 -> K chunks [0,32) and [32,64))
  const _Float16* qrow = qws + ((size_t)(bh*TT + q0 + l15)) * HD;
  const int cb = hl ? 8 : 0;
  const v16h qa0 = cat16(ld8(qrow + cb),      ld8(qrow + cb + 16));
  const v16h qa1 = cat16(ld8(qrow + 32 + cb), ld8(qrow + 32 + cb + 16));

  v8f o[4] = {};
  float mrow[8], lrow[8];
#pragma unroll
  for (int g = 0; g < 8; ++g) { mrow[g] = -1e30f; lrow[g] = 0.0f; }

  stagekv(0, 0);
  wait_async0();
  __syncthreads();

  for (int j = 0; j < jend; j += 32) {
    const int buf = (j >> 5) & 1;
    if (j + 32 < jend) stagekv(j + 32, buf ^ 1);

    if (j < q0 + 16) {   // wave-uniform causal bound: EXEC all-ones inside
      // ---- S = (Q K^T) * 1/sqrt(HD) ----
      v8f s0 = {}, s1 = {};
#pragma unroll
      for (int s = 0; s < 2; ++s) {
        const _Float16* kr = &kt[buf][(s * 16 + l15) * KPAD + (hl ? 8 : 0)];
        v16h kb0 = cat16(ld8(kr),      ld8(kr + 16));
        v16h kb1 = cat16(ld8(kr + 32), ld8(kr + 48));
        v8f a = s ? s1 : s0;
        a = wmma(qa0, kb0, a);
        a = wmma(qa1, kb1, a);
        if (s) s1 = a; else s0 = a;
      }
      // ---- scale + causal mask ----
#pragma unroll
      for (int g = 0; g < 8; ++g) {
        const int qr = q0 + g + (hl ? 8 : 0);
        s0[g] *= 0.125f; s1[g] *= 0.125f;
        if (j + l15      > qr) s0[g] = -1e30f;
        if (j + 16 + l15 > qr) s1[g] = -1e30f;
      }
      // ---- online softmax, P -> per-wave LDS tile ----
#pragma unroll
      for (int g = 0; g < 8; ++g) {
        float v = fmaxf(s0[g], s1[g]);
#pragma unroll
        for (int off = 1; off < 16; off <<= 1)
          v = fmaxf(v, __shfl_xor(v, off, 32));
        const float mnew  = fmaxf(mrow[g], v);
        const float alpha = __expf(mrow[g] - mnew);
        const float p0 = __expf(s0[g] - mnew);
        const float p1 = __expf(s1[g] - mnew);
        float rs = p0 + p1;
#pragma unroll
        for (int off = 1; off < 16; off <<= 1)
          rs += __shfl_xor(rs, off, 32);
        lrow[g] = lrow[g] * alpha + rs;
        mrow[g] = mnew;
#pragma unroll
        for (int t = 0; t < 4; ++t) o[t][g] *= alpha;
        const int ml = g + (hl ? 8 : 0);
        pt[wv][ml*32 + l15]      = (_Float16)p0;
        pt[wv][ml*32 + 16 + l15] = (_Float16)p1;
      }
      // ---- O += P V (P reloaded in A-layout; V^T rows are B-operands) ----
      const _Float16* pr = &pt[wv][l15 * 32 + (hl ? 8 : 0)];
      v16h pa = cat16(ld8(pr), ld8(pr + 16));
#pragma unroll
      for (int t = 0; t < 4; ++t) {
        const _Float16* vr = &vt[buf][(t * 16 + l15) * VPAD + (hl ? 8 : 0)];
        v16h vb = cat16(ld8(vr), ld8(vr + 16));
        o[t] = wmma(pa, vb, o[t]);
      }
    }
    wait_async0();
    __syncthreads();
  }

#pragma unroll
  for (int g = 0; g < 8; ++g) {
    const float inv = 1.0f / lrow[g];
    const int trow  = q0 + g + (hl ? 8 : 0);
    const size_t ro = ((size_t)(bq*TT + trow)) * CC + hh * HD;
#pragma unroll
    for (int t = 0; t < 4; ++t)
      yws[ro + t*16 + l15] = (_Float16)(o[t][g] * inv);
  }
}

// ---------------------------------------------------------------------------
// Kernel 3: out = y @ w_proj^T + b_proj (f32 out). Same staging as kernel 1.
// ---------------------------------------------------------------------------
__global__ __launch_bounds__(128)
void proj_kernel(const _Float16* __restrict__ yws, const _Float16* __restrict__ wh,
                 const float* __restrict__ bias, float* __restrict__ out)
{
  __shared__ _Float16 bt[2][64 * BPAD];

  const int tid  = threadIdx.x;
  const int wv   = tid >> 5;
  const int lane = tid & 31;
  const int hl   = lane >> 4;
  const int l15  = lane & 15;
  const int n0   = blockIdx.x * 64;
  const int m0   = blockIdx.y * 64 + wv * 16;

  auto stage = [&](int k0, int buf) {
    const unsigned base = lds_addr(&bt[buf][0]);
#pragma unroll
    for (int i = 0; i < 2; ++i) {
      const int c = tid + 128 * i;
      const int r = c >> 2, ck = c & 3;
      async_b128(base + (unsigned)(r * BPAD + ck * 8) * 2,
                 wh + (size_t)(n0 + r) * CC + k0 + ck * 8);
    }
  };

  v8f acc[4] = {};
  const _Float16* arow = yws + (size_t)(m0 + l15) * CC;

  stage(0, 0);
  wait_async0();
  __syncthreads();

  for (int k0 = 0; k0 < CC; k0 += 32) {
    const int buf = (k0 >> 5) & 1;
    if (k0 + 32 < CC) stage(k0 + 32, buf ^ 1);

    const int cb = k0 + (hl ? 8 : 0);
    v16h a = cat16(ld8(arow + cb), ld8(arow + cb + 16));
#pragma unroll
    for (int s = 0; s < 4; ++s) {
      const _Float16* br = &bt[buf][(s * 16 + l15) * BPAD + (hl ? 8 : 0)];
      v16h b = cat16(ld8(br), ld8(br + 16));
      acc[s] = wmma(a, b, acc[s]);
    }
    wait_async0();
    __syncthreads();
  }

#pragma unroll
  for (int s = 0; s < 4; ++s) {
    const int n = n0 + s * 16 + l15;
    const float bv = bias[n];
#pragma unroll
    for (int g = 0; g < 8; ++g) {
      const int m = m0 + g + (hl ? 8 : 0);
      out[(size_t)m * CC + n] = acc[s][g] + bv;
    }
  }
}

// ---------------------------------------------------------------------------
extern "C" void kernel_launch(void* const* d_in, const int* in_sizes, int n_in,
                              void* d_out, int out_size, void* d_ws, size_t ws_size,
                              hipStream_t stream) {
  (void)in_sizes; (void)n_in; (void)out_size; (void)ws_size;
  const float* x      = (const float*)d_in[0];
  const float* w_qkv  = (const float*)d_in[1];
  const float* b_qkv  = (const float*)d_in[2];
  const float* w_proj = (const float*)d_in[3];
  const float* b_proj = (const float*)d_in[4];
  float* out = (float*)d_out;

  const size_t nx  = (size_t)NROW * CC;        // 8,388,608
  const size_t nwq = (size_t)3 * CC * CC;      // 3,145,728
  const size_t nwp = (size_t)CC * CC;          // 1,048,576
  const size_t per = (size_t)BB * HH * TT * HD;

  _Float16* ws   = (_Float16*)d_ws;
  _Float16* xh   = ws;                 // also reused as y (f16) after QKV GEMM
  _Float16* wqh  = xh  + nx;
  _Float16* wph  = wqh + nwq;
  _Float16* qws  = wph + nwp;
  _Float16* kws  = qws + per;
  _Float16* vwsT = kws + per;
  _Float16* yws  = xh;                 // alias: x is dead after qkv_kernel

  dim3 blk256(256, 1, 1), blk128(128, 1, 1);
  cvt_kernel<<<dim3((unsigned)(nx  / 2048)), blk256, 0, stream>>>(x,      xh,  (int)nx);
  cvt_kernel<<<dim3((unsigned)(nwq / 2048)), blk256, 0, stream>>>(w_qkv,  wqh, (int)nwq);
  cvt_kernel<<<dim3((unsigned)(nwp / 2048)), blk256, 0, stream>>>(w_proj, wph, (int)nwp);

  qkv_kernel <<<dim3(3*CC/64, NROW/64), blk128, 0, stream>>>(xh, wqh, b_qkv, qws, kws, vwsT);
  attn_kernel<<<dim3(TT/64,   BB*HH),   blk128, 0, stream>>>(qws, kws, vwsT, yws);
  proj_kernel<<<dim3(CC/64,   NROW/64), blk128, 0, stream>>>(yws, wph, b_proj, out);
}
